// OntoGNN_64673617543268
// MI455X (gfx1250) — matrix operand order repaired and checked by reading, not compiled, not run on hardware
//
#include <hip/hip_runtime.h>
#include <cstdint>

#define HDIM 64
#define HEADS 4
#define NV 100000
#define NC 40000
#define NE 60000
#define E_EE 400000
#define E_MAP 250000
#define E_CON 500000

typedef __attribute__((ext_vector_type(16))) __bf16 v16bf;
typedef __attribute__((ext_vector_type(8)))  float  v8f;

// ---- monotone float<->uint key for atomicMax-based segment max ----
__device__ __forceinline__ unsigned fkey(float x) {
    unsigned b = __float_as_uint(x);
    return (b & 0x80000000u) ? ~b : (b | 0x80000000u);
}
__device__ __forceinline__ float unfkey(unsigned k) {
    return (k & 0x80000000u) ? __uint_as_float(k & 0x7fffffffu)
                             : __uint_as_float(~k);
}
#define NEG_INF_KEY 0x007FFFFFu  // fkey(-inf)

__global__ __launch_bounds__(256) void fill_u32_kernel(unsigned* p, long long n, unsigned v) {
    long long i = (long long)blockIdx.x * blockDim.x + threadIdx.x;
    long long stride = (long long)gridDim.x * blockDim.x;
    for (; i < n; i += stride) p[i] = v;
}

#define WMMA_BF16(a, b, c) \
    __builtin_amdgcn_wmma_f32_16x16x32_bf16(false, (a), false, (b), (short)0, (c), false, false)

// ---------------------------------------------------------------------------
// Pack B [K,N] fp32 row-major into per-fragment bf16 layout (zero-padded to
// Kpad = ceil(K/32)*32). Fragment (kb, t) holds the WMMA B operand for
// K-rows [kb*32, kb*32+32) x cols [t*16, t*16+16):
//   lane L: col = t*16 + (L&15); halves i<8 = B[kb*32 + (L<16?0:8) + i][col],
//   halves i>=8 = B[kb*32 + 16 + (L<16?0:8) + (i-8)][col].
// Each lane's 16 halves are stored contiguously (32B) -> single v16bf load.
// ---------------------------------------------------------------------------
__global__ __launch_bounds__(256) void pack_b_kernel(
        const float* __restrict__ B, __bf16* __restrict__ PB, int K, int N) {
    int Kpad = (K + 31) & ~31;
    int nfrag = (Kpad >> 5) * (N >> 4);
    int idx = blockIdx.x * blockDim.x + threadIdx.x;
    if (idx >= nfrag * 32) return;
    int frag = idx >> 5, lane = idx & 31;
    int ntile = N >> 4;
    int kb = frag / ntile, t = frag % ntile;
    int col = (t << 4) + (lane & 15);
    int kbase = (kb << 5) + (lane < 16 ? 0 : 8);
    __bf16* out = PB + (size_t)idx * 16;
#pragma unroll
    for (int i = 0; i < 16; i++) {
        int k = kbase + (i < 8 ? i : i + 8);
        out[i] = (k < K) ? (__bf16)B[(size_t)k * N + col] : (__bf16)0.0f;
    }
}

// A fragment (16-bit 16x32 layout): lanes 0-15 row=lane, halves 0-7 = K{0..7},
// halves 8-15 = K{16..23}; lanes 16-31 same rows, K{8..15} and K{24..31}.
__device__ __forceinline__ v16bf load_a_frag(const float* __restrict__ Arow,
                                             int ka0, int ka1) {
    float4 f0 = *(const float4*)(Arow + ka0);
    float4 f1 = *(const float4*)(Arow + ka0 + 4);
    float4 f2 = *(const float4*)(Arow + ka1);
    float4 f3 = *(const float4*)(Arow + ka1 + 4);
    v16bf a;
    a[0]  = (__bf16)f0.x; a[1]  = (__bf16)f0.y; a[2]  = (__bf16)f0.z; a[3]  = (__bf16)f0.w;
    a[4]  = (__bf16)f1.x; a[5]  = (__bf16)f1.y; a[6]  = (__bf16)f1.z; a[7]  = (__bf16)f1.w;
    a[8]  = (__bf16)f2.x; a[9]  = (__bf16)f2.y; a[10] = (__bf16)f2.z; a[11] = (__bf16)f2.w;
    a[12] = (__bf16)f3.x; a[13] = (__bf16)f3.y; a[14] = (__bf16)f3.z; a[15] = (__bf16)f3.w;
    return a;
}

// ---------------------------------------------------------------------------
// Fast path: C[M,N] = act([A0|A1] @ B + bias), requires K0%32==0 && K1%32==0,
// N%64==0, M%16==0. One wave computes a 16x64 strip (4 WMMA accumulators
// sharing one A fragment per 32-deep K step). B comes pre-packed (PB).
// C/D frag: VGPR r -> row r (lanes<16) / r+8 (lanes>=16), col = lane&15.
// ---------------------------------------------------------------------------
__global__ __launch_bounds__(256) void gemm_bf16_wmma_aligned_kernel(
        const float* __restrict__ A0, int K0,
        const float* __restrict__ A1, int K1,
        const __bf16* __restrict__ PB, const float* __restrict__ bias,
        float* __restrict__ C, int M, int N, int relu) {
    int wave = (int)((blockIdx.x * blockDim.x + threadIdx.x) >> 5);
    int lane = threadIdx.x & 31;
    int ngrp = N >> 6;
    int total = (M >> 4) * ngrp;
    if (wave >= total) return;
    int tM = wave / ngrp, tG = wave % ngrp;
    int row  = (tM << 4) + (lane & 15);
    int col0 = (tG << 6) + (lane & 15);
    bool lo = lane < 16;
    int K = K0 + K1;
    int ntile = N >> 4;
    v8f z = {0.f, 0.f, 0.f, 0.f, 0.f, 0.f, 0.f, 0.f};
    v8f acc0 = z, acc1 = z, acc2 = z, acc3 = z;
    for (int k0 = 0; k0 < K; k0 += 32) {
        const float* Arow;
        int kl;
        if (k0 < K0) { Arow = A0 + (size_t)row * K0; kl = k0; }
        else         { Arow = A1 + (size_t)row * K1; kl = k0 - K0; }
        v16bf a = load_a_frag(Arow, kl + (lo ? 0 : 8), kl + (lo ? 16 : 24));
        // four contiguous 32B fragment loads from packed B
        const v16bf* pb = (const v16bf*)(PB +
            (((size_t)(k0 >> 5) * ntile + (tG << 2)) * 32 + lane) * 16);
        v16bf b0 = pb[0 * 32];
        v16bf b1 = pb[1 * 32];
        v16bf b2 = pb[2 * 32];
        v16bf b3 = pb[3 * 32];
        acc0 = WMMA_BF16(a, b0, acc0);
        acc1 = WMMA_BF16(a, b1, acc1);
        acc2 = WMMA_BF16(a, b2, acc2);
        acc3 = WMMA_BF16(a, b3, acc3);
    }
    int r0 = (tM << 4) + (lo ? 0 : 8);
    v8f accs[4] = {acc0, acc1, acc2, acc3};
#pragma unroll
    for (int t = 0; t < 4; t++) {
        int col = col0 + t * 16;
        float bb = bias ? bias[col] : 0.f;
#pragma unroll
        for (int r = 0; r < 8; r++) {
            float v = accs[t][r] + bb;
            if (relu) v = fmaxf(v, 0.f);
            C[(size_t)(r0 + r) * N + col] = v;
        }
    }
}

// Slow path for ragged K (19, 5): clamped A loads; B comes pre-packed (padded).
__global__ __launch_bounds__(256) void gemm_bf16_wmma_padded_kernel(
        const float* __restrict__ A, int K,
        const __bf16* __restrict__ PB, const float* __restrict__ bias,
        float* __restrict__ C, int M, int N, int relu) {
    int wave = (int)((blockIdx.x * blockDim.x + threadIdx.x) >> 5);
    int lane = threadIdx.x & 31;
    int ngrp = N >> 6;
    int total = (M >> 4) * ngrp;
    if (wave >= total) return;
    int tM = wave / ngrp, tG = wave % ngrp;
    int row  = (tM << 4) + (lane & 15);
    int col0 = (tG << 6) + (lane & 15);
    bool lo = lane < 16;
    int Kpad = (K + 31) & ~31;
    int ntile = N >> 4;
    v8f z = {0.f, 0.f, 0.f, 0.f, 0.f, 0.f, 0.f, 0.f};
    v8f acc0 = z, acc1 = z, acc2 = z, acc3 = z;
    const float* Arow = A + (size_t)row * K;
    for (int k0 = 0; k0 < Kpad; k0 += 32) {
        int ka0 = k0 + (lo ? 0 : 8);
        int ka1 = k0 + (lo ? 16 : 24);
        v16bf a;
#pragma unroll
        for (int i = 0; i < 8; i++) {
            int k = ka0 + i;
            float va = Arow[k < K ? k : 0];
            a[i] = (__bf16)(k < K ? va : 0.f);
            k = ka1 + i;
            va = Arow[k < K ? k : 0];
            a[8 + i] = (__bf16)(k < K ? va : 0.f);
        }
        const v16bf* pb = (const v16bf*)(PB +
            (((size_t)(k0 >> 5) * ntile + (tG << 2)) * 32 + lane) * 16);
        v16bf b0 = pb[0 * 32];
        v16bf b1 = pb[1 * 32];
        v16bf b2 = pb[2 * 32];
        v16bf b3 = pb[3 * 32];
        acc0 = WMMA_BF16(a, b0, acc0);
        acc1 = WMMA_BF16(a, b1, acc1);
        acc2 = WMMA_BF16(a, b2, acc2);
        acc3 = WMMA_BF16(a, b3, acc3);
    }
    int r0 = (tM << 4) + (lo ? 0 : 8);
    v8f accs[4] = {acc0, acc1, acc2, acc3};
#pragma unroll
    for (int t = 0; t < 4; t++) {
        int col = col0 + t * 16;
        float bb = bias ? bias[col] : 0.f;
#pragma unroll
        for (int r = 0; r < 8; r++) {
            float v = accs[t][r] + bb;
            if (relu) v = fmaxf(v, 0.f);
            C[(size_t)(r0 + r) * N + col] = v;
        }
    }
}

// asrc[n,h] = sum_d HS[n,h,d] * avec[h,d]; one wave per node (wave32: 2 elems/lane)
__global__ __launch_bounds__(256) void att_dot_kernel(
        const float* __restrict__ Hs, const float* __restrict__ avec,
        float* __restrict__ out, int Nn) {
    int wave = (int)((blockIdx.x * blockDim.x + threadIdx.x) >> 5);
    int lane = threadIdx.x & 31;
    if (wave >= Nn) return;
    const float* hp = Hs + (size_t)wave * (HEADS * HDIM);
#pragma unroll
    for (int h = 0; h < HEADS; h++) {
        float p = hp[h * HDIM + lane] * avec[h * HDIM + lane]
                + hp[h * HDIM + lane + 32] * avec[h * HDIM + lane + 32];
#pragma unroll
        for (int off = 16; off; off >>= 1) p += __shfl_xor(p, off, 32);
        if (lane == 0) out[wave * HEADS + h] = p;
    }
}

// e = leaky_relu(asrc[src]+adst[dst]); store; segment-max via keyed atomicMax
__global__ __launch_bounds__(256) void gat_edge_max_kernel(
        const int* __restrict__ src, const int* __restrict__ dst, int E, int selfloop,
        const float* __restrict__ asrc, const float* __restrict__ adst,
        float* __restrict__ ebuf, unsigned* __restrict__ segmax) {
    long long i = (long long)blockIdx.x * blockDim.x + threadIdx.x;
    if (i >= (long long)E * HEADS) return;
    int e = (int)(i / HEADS), h = (int)(i % HEADS);
    int s = selfloop ? e : src[e];
    int d = selfloop ? e : dst[e];
    float v = asrc[s * HEADS + h] + adst[d * HEADS + h];
    v = v > 0.f ? v : 0.2f * v;
    ebuf[i] = v;
    atomicMax(&segmax[d * HEADS + h], fkey(v));
}

// ex = exp(e - max[dst]); segment-sum; generic in head count
__global__ __launch_bounds__(256) void edge_expsum_kernel(
        const int* __restrict__ dst, int E, int selfloop,
        float* __restrict__ ebuf, const unsigned* __restrict__ segmax,
        float* __restrict__ segsum, int heads) {
    long long i = (long long)blockIdx.x * blockDim.x + threadIdx.x;
    if (i >= (long long)E * heads) return;
    int e = (int)(i / heads), h = (int)(i % heads);
    int d = selfloop ? e : dst[e];
    float m = unfkey(segmax[d * heads + h]);
    float ex = __expf(ebuf[i] - m);
    ebuf[i] = ex;
    atomicAdd(&segsum[d * heads + h], ex);
}

// OUT[dst,h,:] += HS[src,h,:] * alpha; one wave per edge, 2 floats per lane
__global__ __launch_bounds__(256) void gat_scatter_kernel(
        const int* __restrict__ src, const int* __restrict__ dst, int E, int selfloop,
        const float* __restrict__ Hs, const float* __restrict__ ebuf,
        const float* __restrict__ segsum, float* __restrict__ outacc) {
    int wave = (int)((blockIdx.x * blockDim.x + threadIdx.x) >> 5);
    int lane = threadIdx.x & 31;
    if (wave >= E) return;
    int s = selfloop ? wave : src[wave];
    int d = selfloop ? wave : dst[wave];
    const float* hp = Hs + (size_t)s * (HEADS * HDIM);
    float* op = outacc + (size_t)d * (HEADS * HDIM);
#pragma unroll
    for (int h = 0; h < HEADS; h++) {
        float alpha = ebuf[(size_t)wave * HEADS + h] / (segsum[d * HEADS + h] + 1e-16f);
        atomicAdd(&op[h * HDIM + lane],      hp[h * HDIM + lane] * alpha);
        atomicAdd(&op[h * HDIM + lane + 32], hp[h * HDIM + lane + 32] * alpha);
    }
}

// h_out = relu(mean_heads(OUT) + bias) + resid   (resid == h_out allowed, elementwise)
__global__ __launch_bounds__(256) void gat_finalize_kernel(
        const float* __restrict__ outacc, const float* __restrict__ bias,
        const float* __restrict__ resid, float* __restrict__ hout, int Nn) {
    long long i = (long long)blockIdx.x * blockDim.x + threadIdx.x;
    if (i >= (long long)Nn * HDIM) return;
    int n = (int)(i >> 6), c = (int)(i & 63);
    const float* op = outacc + (size_t)n * (HEADS * HDIM);
    float v = 0.25f * (op[c] + op[HDIM + c] + op[2 * HDIM + c] + op[3 * HDIM + c]) + bias[c];
    v = fmaxf(v, 0.f);
    hout[i] = v + resid[i];
}

// stage 2: e = dot(Q[var],K[ent])/8; keyed segment max over var
__global__ __launch_bounds__(256) void qk_edge_max_kernel(
        const int* __restrict__ evar, const int* __restrict__ eent, int E,
        const float* __restrict__ Q, const float* __restrict__ Km,
        float* __restrict__ ebuf, unsigned* __restrict__ segmax) {
    int wave = (int)((blockIdx.x * blockDim.x + threadIdx.x) >> 5);
    int lane = threadIdx.x & 31;
    if (wave >= E) return;
    int qv = evar[wave], ke = eent[wave];
    const float* qp = Q + (size_t)qv * HDIM;
    const float* kp = Km + (size_t)ke * HDIM;
    float p = qp[lane] * kp[lane] + qp[lane + 32] * kp[lane + 32];
#pragma unroll
    for (int off = 16; off; off >>= 1) p += __shfl_xor(p, off, 32);
    if (lane == 0) {
        float v = p * 0.125f;  // 1/sqrt(64)
        ebuf[wave] = v;
        atomicMax(&segmax[qv], fkey(v));
    }
}

// stage 2 scatter: agg[var,:] += V[ent,:] * ex / max(sum, 1e-9)
__global__ __launch_bounds__(256) void map_scatter_kernel(
        const int* __restrict__ evar, const int* __restrict__ eent, int E,
        const float* __restrict__ V, const float* __restrict__ ebuf,
        const float* __restrict__ segsum, float* __restrict__ agg) {
    int wave = (int)((blockIdx.x * blockDim.x + threadIdx.x) >> 5);
    int lane = threadIdx.x & 31;
    if (wave >= E) return;
    int dv = evar[wave], se = eent[wave];
    float alpha = ebuf[wave] / fmaxf(segsum[dv], 1e-9f);
    atomicAdd(&agg[(size_t)dv * HDIM + lane],      V[(size_t)se * HDIM + lane] * alpha);
    atomicAdd(&agg[(size_t)dv * HDIM + lane + 32], V[(size_t)se * HDIM + lane + 32] * alpha);
}

// final score: out[n] = dot(h1[n,:], w2) + b2
__global__ __launch_bounds__(256) void score_kernel(
        const float* __restrict__ h1, const float* __restrict__ w2,
        const float* __restrict__ b2, float* __restrict__ out, int Nn) {
    int wave = (int)((blockIdx.x * blockDim.x + threadIdx.x) >> 5);
    int lane = threadIdx.x & 31;
    if (wave >= Nn) return;
    const float* hp = h1 + (size_t)wave * HDIM;
    float p = hp[lane] * w2[lane] + hp[lane + 32] * w2[lane + 32];
#pragma unroll
    for (int off = 16; off; off >>= 1) p += __shfl_xor(p, off, 32);
    if (lane == 0) out[wave] = p + b2[0];
}

extern "C" void kernel_launch(void* const* d_in, const int* in_sizes, int n_in,
                              void* d_out, int out_size, void* d_ws, size_t ws_size,
                              hipStream_t stream) {
    (void)in_sizes; (void)n_in; (void)out_size; (void)ws_size;
    const float* x_var = (const float*)d_in[0];
    const float* x_con = (const float*)d_in[1];
    const float* x_ent = (const float*)d_in[2];
    const int* ei      = (const int*)d_in[3];
    const int* ee_src  = ei;
    const int* ee_dst  = ei + E_EE;
    const int* map_var = (const int*)d_in[4];
    const int* map_ent = (const int*)d_in[5];
    const int* con_var = (const int*)d_in[6];
    const int* con_con = (const int*)d_in[7];
    int p = 8;
    const float* pv_W = (const float*)d_in[p++]; const float* pv_b = (const float*)d_in[p++];
    const float* pc_W = (const float*)d_in[p++]; const float* pc_b = (const float*)d_in[p++];
    const float* pe_W = (const float*)d_in[p++]; const float* pe_b = (const float*)d_in[p++];
    const float* sem_W = (const float*)d_in[p++]; const float* sem_as = (const float*)d_in[p++];
    const float* sem_ad = (const float*)d_in[p++]; const float* sem_b = (const float*)d_in[p++];
    const float* q_W = (const float*)d_in[p++]; const float* k_W = (const float*)d_in[p++];
    const float* v_W = (const float*)d_in[p++]; const float* v_b = (const float*)d_in[p++];
    const float* fuse_W = (const float*)d_in[p++]; const float* fuse_b = (const float*)d_in[p++];
    const float *g_Ws[4], *g_Wd[4], *g_as[4], *g_ad[4], *g_b[4];  // v2c[0],v2c[1],c2v[0],c2v[1]
    for (int l = 0; l < 4; l++) {
        g_Ws[l] = (const float*)d_in[p++]; g_Wd[l] = (const float*)d_in[p++];
        g_as[l] = (const float*)d_in[p++]; g_ad[l] = (const float*)d_in[p++];
        g_b[l]  = (const float*)d_in[p++];
    }
    const float* s1_W = (const float*)d_in[p++]; const float* s1_b = (const float*)d_in[p++];
    const float* s2_W = (const float*)d_in[p++]; const float* s2_b = (const float*)d_in[p++];

    // ---- workspace carve (floats; every region a multiple of 4 floats) ----
    float* ws = (float*)d_ws;
    size_t o = 0;
    float* HV  = ws + o; o += (size_t)NV * HDIM;
    float* HV2 = ws + o; o += (size_t)NV * HDIM;
    float* HC  = ws + o; o += (size_t)NC * HDIM;
    float* HE  = ws + o; o += (size_t)NE * HDIM;
    float* BUFA = ws + o; o += (size_t)NV * HEADS * HDIM;  // HS (src transformed)
    float* BUFB = ws + o; o += (size_t)NV * HEADS * HDIM;  // HD (dst transformed) / K,V
    float* BUFO = ws + o; o += (size_t)NV * HEADS * HDIM;  // attention accumulator / agg
    float* ASRC = ws + o; o += (size_t)NV * HEADS;
    float* ADST = ws + o; o += (size_t)NV * HEADS;
    unsigned* SMAX = (unsigned*)(ws + o); o += (size_t)NV * HEADS;
    float* SSUM = ws + o; o += (size_t)NV * HEADS;
    float* EBUF = ws + o; o += (size_t)E_CON * HEADS;
    __bf16* PB = (__bf16*)(ws + o); o += 16384;  // packed B: up to 128x256 bf16

    auto cdiv = [](long long a, long long b) { return (int)((a + b - 1) / b); };
    auto gemm = [&](const float* A0, int K0, const float* A1, int K1,
                    const float* B, const float* bias, float* C, int M, int N, int relu) {
        int K = K0 + K1;
        int Kpad = (K + 31) & ~31;
        int packThreads = (Kpad >> 5) * (N >> 4) * 32;
        pack_b_kernel<<<cdiv(packThreads, 256), 256, 0, stream>>>(B, PB, K, N);
        long long waves = (long long)(M / 16) * (N / 64);
        if ((K0 & 31) == 0)
            gemm_bf16_wmma_aligned_kernel<<<cdiv(waves * 32, 256), 256, 0, stream>>>(
                A0, K0, A1, K1, PB, bias, C, M, N, relu);
        else
            gemm_bf16_wmma_padded_kernel<<<cdiv(waves * 32, 256), 256, 0, stream>>>(
                A0, K0, PB, bias, C, M, N, relu);
    };
    auto fill = [&](void* ptr, long long n, unsigned v) {
        int b = cdiv(n, 256); if (b > 32768) b = 32768;
        fill_u32_kernel<<<b, 256, 0, stream>>>((unsigned*)ptr, n, v);
    };

    // Generic multi-head GAT pass (PyG GATConv, heads=4, concat=False)
    auto run_gat = [&](const float* xsrc, int Ns, const float* Ws,
                       const float* xdst, int Nd, const float* Wd,
                       const float* a_s, const float* a_d, const float* bias,
                       const int* src, const int* dst, int E, int selfloops,
                       float* hdst_inout) {
        gemm(xsrc, HDIM, nullptr, 0, Ws, nullptr, BUFA, Ns, HEADS * HDIM, 0);
        const float* HDuse;
        if (Ws == Wd && xsrc == xdst) { HDuse = BUFA; }
        else { gemm(xdst, HDIM, nullptr, 0, Wd, nullptr, BUFB, Nd, HEADS * HDIM, 0); HDuse = BUFB; }
        att_dot_kernel<<<cdiv((long long)Ns * 32, 256), 256, 0, stream>>>(BUFA, a_s, ASRC, Ns);
        att_dot_kernel<<<cdiv((long long)Nd * 32, 256), 256, 0, stream>>>(HDuse, a_d, ADST, Nd);
        fill(SMAX, (long long)Nd * HEADS, NEG_INF_KEY);
        fill(SSUM, (long long)Nd * HEADS, 0u);
        fill(BUFO, (long long)Nd * HEADS * HDIM, 0u);
        gat_edge_max_kernel<<<cdiv((long long)E * HEADS, 256), 256, 0, stream>>>(
            src, dst, E, 0, ASRC, ADST, EBUF, SMAX);
        if (selfloops)
            gat_edge_max_kernel<<<cdiv((long long)Nd * HEADS, 256), 256, 0, stream>>>(
                nullptr, nullptr, Nd, 1, ASRC, ADST, EBUF + (size_t)E * HEADS, SMAX);
        edge_expsum_kernel<<<cdiv((long long)E * HEADS, 256), 256, 0, stream>>>(
            dst, E, 0, EBUF, SMAX, SSUM, HEADS);
        if (selfloops)
            edge_expsum_kernel<<<cdiv((long long)Nd * HEADS, 256), 256, 0, stream>>>(
                nullptr, Nd, 1, EBUF + (size_t)E * HEADS, SMAX, SSUM, HEADS);
        gat_scatter_kernel<<<cdiv((long long)E * 32, 256), 256, 0, stream>>>(
            src, dst, E, 0, BUFA, EBUF, SSUM, BUFO);
        if (selfloops)
            gat_scatter_kernel<<<cdiv((long long)Nd * 32, 256), 256, 0, stream>>>(
                nullptr, nullptr, Nd, 1, BUFA, EBUF + (size_t)E * HEADS, SSUM, BUFO);
        gat_finalize_kernel<<<cdiv((long long)Nd * HDIM, 256), 256, 0, stream>>>(
            BUFO, bias, hdst_inout, hdst_inout, Nd);
    };

    // ---- Stage 0: input projections (K=19/5 use padded path, K=128 fast path) ----
    gemm(x_var, 19, nullptr, 0, pv_W, pv_b, HV, NV, HDIM, 1);
    gemm(x_con, 5, nullptr, 0, pc_W, pc_b, HC, NC, HDIM, 1);
    gemm(x_ent, 128, nullptr, 0, pe_W, pe_b, HE, NE, HDIM, 1);

    // ---- Stage 1: entity-entity GAT with self loops + residual ----
    run_gat(HE, NE, sem_W, HE, NE, sem_W, sem_as, sem_ad, sem_b,
            ee_src, ee_dst, E_EE, 1, HE);

    // ---- Stage 2: sparse QKV attention entity -> variable ----
    float* Qb = BUFA;
    float* Kb = BUFB;
    float* Vb = BUFB + (size_t)NE * HDIM;
    float* AGG = BUFO;
    gemm(HV, HDIM, nullptr, 0, q_W, nullptr, Qb, NV, HDIM, 0);
    gemm(HE, HDIM, nullptr, 0, k_W, nullptr, Kb, NE, HDIM, 0);
    gemm(HE, HDIM, nullptr, 0, v_W, v_b, Vb, NE, HDIM, 0);
    fill(SMAX, NV, NEG_INF_KEY);
    fill(SSUM, NV, 0u);
    fill(AGG, (long long)NV * HDIM, 0u);
    qk_edge_max_kernel<<<cdiv((long long)E_MAP * 32, 256), 256, 0, stream>>>(
        map_var, map_ent, E_MAP, Qb, Kb, EBUF, SMAX);
    edge_expsum_kernel<<<cdiv((long long)E_MAP, 256), 256, 0, stream>>>(
        map_var, E_MAP, 0, EBUF, SMAX, SSUM, 1);
    map_scatter_kernel<<<cdiv((long long)E_MAP * 32, 256), 256, 0, stream>>>(
        map_var, map_ent, E_MAP, Vb, EBUF, SSUM, AGG);
    // fuse: relu(concat([hv, agg]) @ fuse_W + fuse_b) -> HV2 (dual-A GEMM, K0=K1=64)
    gemm(HV, HDIM, AGG, HDIM, fuse_W, fuse_b, HV2, NV, HDIM, 1);
    float* hv = HV2;

    // ---- Stage 3: two rounds of bipartite GAT with residuals ----
    for (int l = 0; l < 2; l++) {
        run_gat(hv, NV, g_Ws[l], HC, NC, g_Wd[l], g_as[l], g_ad[l], g_b[l],
                con_var, con_con, E_CON, 0, HC);
        run_gat(HC, NC, g_Ws[2 + l], hv, NV, g_Wd[2 + l], g_as[2 + l], g_ad[2 + l], g_b[2 + l],
                con_con, con_var, E_CON, 0, hv);
    }

    // ---- Stage 4: scoring head ----
    gemm(hv, HDIM, nullptr, 0, s1_W, s1_b, BUFA, NV, HDIM, 1);
    score_kernel<<<cdiv((long long)NV * 32, 256), 256, 0, stream>>>(
        BUFA, s2_W, s2_b, (float*)d_out, NV);
}
